// PNA_27857157882092
// MI455X (gfx1250) — matrix operand order repaired and checked by reading, not compiled, not run on hardware
//
#include <hip/hip_runtime.h>
#include <math.h>

// Problem constants (from reference)
#define NN   50000
#define EE   400000
#define EPP  40000
#define FF   64
#define EDD  32
#define HH   128
#define LL   2

typedef _Float16 v16h __attribute__((ext_vector_type(16)));
typedef float    v8f  __attribute__((ext_vector_type(8)));
typedef _Float16 h2   __attribute__((ext_vector_type(2)));

// ---------------------------------------------------------------------------
// WMMA fragment helpers (layouts per CDNA5 ISA 7.12.2, wave32)
//   A 16x32 f16 : lane L(0-15) row M=L, VGPR i<4 -> K=2i+8h, i>=4 -> K=2i+8h+8
//   B 32x16 f16 : lane L col N=L&15, VGPR i<4 -> K=2i+8h, i>=4 -> K=2i+8h+8
//   C 16x16 f32 : VGPR r -> row M=r+8h, col N=lane&15
// ---------------------------------------------------------------------------
__device__ __forceinline__ v16h frag_a_f32(const float* __restrict__ row, int k0, int h) {
  v16h a;
#pragma unroll
  for (int i = 0; i < 8; ++i) {
    int kb = k0 + 2 * i + 8 * h + ((i & 4) ? 8 : 0);
    float2 p = *(const float2*)(row + kb);
    a[2 * i]     = (_Float16)p.x;
    a[2 * i + 1] = (_Float16)p.y;
  }
  return a;
}

__device__ __forceinline__ v16h frag_a_f32s(const float* __restrict__ row, int k0, int h, float sc) {
  v16h a;
#pragma unroll
  for (int i = 0; i < 8; ++i) {
    int kb = k0 + 2 * i + 8 * h + ((i & 4) ? 8 : 0);
    float2 p = *(const float2*)(row + kb);
    a[2 * i]     = (_Float16)(p.x * sc);
    a[2 * i + 1] = (_Float16)(p.y * sc);
  }
  return a;
}

__device__ __forceinline__ v16h frag_a_f16(const _Float16* __restrict__ row, int k0, int h) {
  v16h a;
#pragma unroll
  for (int i = 0; i < 8; ++i) {
    int kb = k0 + 2 * i + 8 * h + ((i & 4) ? 8 : 0);
    h2 p = *(const h2*)(row + kb);
    a[2 * i]     = p.x;
    a[2 * i + 1] = p.y;
  }
  return a;
}

// weights stored N-major: wt[n*K + k]; pairs (2i,2i+1) contiguous
template <int K>
__device__ __forceinline__ v16h frag_b(const _Float16* __restrict__ wtrow, int k0, int h) {
  v16h b;
#pragma unroll
  for (int i = 0; i < 8; ++i) {
    int kb = k0 + 2 * i + 8 * h + ((i & 4) ? 8 : 0);
    h2 p = *(const h2*)(wtrow + kb);
    b[2 * i]     = p.x;
    b[2 * i + 1] = p.y;
  }
  return b;
}

#define WMMA_F16(C, A, B) \
  __builtin_amdgcn_wmma_f32_16x16x32_f16(false, (A), false, (B), (short)0, (C), false, false)

// One K-step for a 16x128 output strip: preload all 8 B fragments into
// distinct registers (loads issue as one clause, latency covered once),
// then stream 8 back-to-back WMMAs (shared A, independent D -> no hazard NOPs).
template <int K>
__device__ __forceinline__ void wmma_step(v8f (&acc)[8], v16h a,
                                          const _Float16* __restrict__ wt,
                                          int l, int k0, int h) {
  v16h bf[8];
#pragma unroll
  for (int nt = 0; nt < 8; ++nt)
    bf[nt] = frag_b<K>(wt + (long long)(nt * 16 + l) * K, k0, h);
#pragma unroll
  for (int nt = 0; nt < 8; ++nt)
    acc[nt] = WMMA_F16(acc[nt], a, bf[nt]);
}

__device__ __forceinline__ void acc_bias(v8f (&acc)[8], const float* __restrict__ bias, int l) {
#pragma unroll
  for (int nt = 0; nt < 8; ++nt) {
    float bv = bias[nt * 16 + l];
#pragma unroll
    for (int r = 0; r < 8; ++r) acc[nt][r] = bv;
  }
}

// float atomic min/max via signed/unsigned int atomics (IEEE ordering trick)
__device__ __forceinline__ void atomicMaxF(float* a, float v) {
  if (v >= 0.f) atomicMax((int*)a, __float_as_int(v));
  else          atomicMin((unsigned int*)a, __float_as_uint(v));
}
__device__ __forceinline__ void atomicMinF(float* a, float v) {
  if (v >= 0.f) atomicMin((int*)a, __float_as_int(v));
  else          atomicMax((unsigned int*)a, __float_as_uint(v));
}

// ---------------------------------------------------------------------------
// Utility kernels
// ---------------------------------------------------------------------------
__global__ void k_fillf(float* p, long long n, float v) {
  long long i = (long long)blockIdx.x * blockDim.x + threadIdx.x;
  long long st = (long long)gridDim.x * blockDim.x;
  for (; i < n; i += st) p[i] = v;
}

// convert fp32 [L][K][128] weights -> fp16 N-major [L][128][K]
__global__ void k_cvt(const float* __restrict__ w, _Float16* __restrict__ wt,
                      int K, long long total) {
  long long i = (long long)blockIdx.x * blockDim.x + threadIdx.x;
  long long st = (long long)gridDim.x * blockDim.x;
  for (; i < total; i += st) {
    int n = (int)(i % 128);
    long long t = i / 128;
    int k = (int)(t % K);
    long long layer = t / K;
    wt[layer * (long long)128 * K + (long long)n * K + k] = (_Float16)w[i];
  }
}

// stats layout per node: [0:128)=sum -> mean, [128:256)=min, [256:384)=max, [384:512)=sumsq -> std
__global__ void k_init_stats(float* s) {
  long long total = (long long)NN * 512;
  long long i = (long long)blockIdx.x * blockDim.x + threadIdx.x;
  long long st = (long long)gridDim.x * blockDim.x;
  for (; i < total; i += st) {
    int g = (int)((i >> 7) & 3);
    s[i] = (g == 1) ? __builtin_inff() : (g == 2) ? -__builtin_inff() : 0.f;
  }
}

__global__ void k_deg(const int* __restrict__ dst, float* deg) {
  int i = blockIdx.x * blockDim.x + threadIdx.x;
  if (i < EE) atomicAdd(deg + dst[i], 1.0f);
}

__global__ void k_nodestats(const float* __restrict__ deg, const float* __restrict__ adl_p,
                            float* amp, float* att) {
  int i = blockIdx.x * blockDim.x + threadIdx.x;
  if (i < NN) {
    float degc = fmaxf(deg[i], 1.f);
    float lg = logf(degc + 1.f);
    float adl = *adl_p;
    amp[i] = lg / adl;
    att[i] = adl / lg;
  }
}

// ---------------------------------------------------------------------------
// Generic input GEMM: C[M x 128] = A[M x K](f32) @ Wt(f16) + bias, out f32 or f16
// one wave computes a 16x128 strip
// ---------------------------------------------------------------------------
template <int K>
__global__ __launch_bounds__(256)
void k_gemm_in(const float* __restrict__ A, const _Float16* __restrict__ wt,
               const float* __restrict__ bias, float* outF, _Float16* outH, int Mtiles) {
  int wave = (int)((blockIdx.x * blockDim.x + threadIdx.x) >> 5);
  int lane = threadIdx.x & 31;
  if (wave >= Mtiles) return;
  int h = lane >> 4, l = lane & 15;
  int m0 = wave * 16;
  const float* arow = A + (long long)(m0 + l) * K;
  v8f acc[8];
  acc_bias(acc, bias, l);
#pragma unroll
  for (int k0 = 0; k0 < K; k0 += 32) {
    v16h a = frag_a_f32(arow, k0, h);
    wmma_step<K>(acc, a, wt, l, k0, h);
  }
#pragma unroll
  for (int nt = 0; nt < 8; ++nt)
#pragma unroll
    for (int r = 0; r < 8; ++r) {
      long long idx = (long long)(m0 + r + 8 * h) * 128 + nt * 16 + l;
      if (outF) outF[idx] = acc[nt][r];
      else      outH[idx] = (_Float16)acc[nt][r];
    }
}

// ---------------------------------------------------------------------------
// Fused edge message: e_enc = ea@eenc + b ; h = [x[dst] x[src] e_enc]@pre + b ;
// scatter h into stats (sum / min / max / sumsq) with atomics.
// ---------------------------------------------------------------------------
__global__ __launch_bounds__(256)
void k_message(const float* __restrict__ x, const _Float16* __restrict__ ea,
               const int* __restrict__ src, const int* __restrict__ dst,
               const _Float16* __restrict__ wt_e, const float* __restrict__ b_e,
               const _Float16* __restrict__ wt_p, const float* __restrict__ b_p,
               float* stats, int Etiles) {
  __shared__ _Float16 lds[8][16 * 128];
  int wslot = threadIdx.x >> 5;
  int wave = (int)((blockIdx.x * blockDim.x + threadIdx.x) >> 5);
  int lane = threadIdx.x & 31;
  if (wave >= Etiles) return;
  int h = lane >> 4, l = lane & 15;
  int e0 = wave * 16;
  int my = e0 + l;

  // --- stage 1: e_enc tile (16x128) = ea_tile @ eenc_w + b ---
  const _Float16* earow = ea + (long long)my * 128;
  v8f acc[8];
  acc_bias(acc, b_e, l);
#pragma unroll
  for (int k0 = 0; k0 < 128; k0 += 32) {
    v16h a = frag_a_f16(earow, k0, h);
    wmma_step<128>(acc, a, wt_e, l, k0, h);
  }
  _Float16* L = lds[wslot];
#pragma unroll
  for (int nt = 0; nt < 8; ++nt)
#pragma unroll
    for (int r = 0; r < 8; ++r)
      L[(r + 8 * h) * 128 + nt * 16 + l] = (_Float16)acc[nt][r];
  // DS pipe is in-order per wave: staging writes land before reads below.

  // --- stage 2: h tile = [x[dst] | x[src] | e_enc] @ pre_w + b ---
  const float* xd = x + (long long)dst[my] * 128;
  const float* xs = x + (long long)src[my] * 128;
  const _Float16* ee = L + l * 128;
  v8f hacc[8];
  acc_bias(hacc, b_p, l);
#pragma unroll
  for (int k0 = 0; k0 < 384; k0 += 32) {
    v16h a;
    if (k0 < 128)      a = frag_a_f32(xd, k0, h);
    else if (k0 < 256) a = frag_a_f32(xs, k0 - 128, h);
    else               a = frag_a_f16(ee, k0 - 256, h);
    wmma_step<384>(hacc, a, wt_p, l, k0, h);
  }

  // --- stage 3: scatter-reduce into stats ---
  int nd[8];
#pragma unroll
  for (int r = 0; r < 8; ++r) nd[r] = dst[e0 + r + 8 * h];
#pragma unroll
  for (int nt = 0; nt < 8; ++nt) {
    int ch = nt * 16 + l;
#pragma unroll
    for (int r = 0; r < 8; ++r) {
      float v = hacc[nt][r];
      float* sb = stats + (long long)nd[r] * 512;
      atomicAdd(sb + ch, v);
      atomicMinF(sb + 128 + ch, v);
      atomicMaxF(sb + 256 + ch, v);
      atomicAdd(sb + 384 + ch, v * v);
    }
  }
}

// in-place: sum->mean, sumsq->std, mask min/max by has_in
__global__ void k_aggfinish(float* stats, const float* __restrict__ deg) {
  long long total = (long long)NN * 128;
  long long i = (long long)blockIdx.x * blockDim.x + threadIdx.x;
  long long st = (long long)gridDim.x * blockDim.x;
  for (; i < total; i += st) {
    int node = (int)(i >> 7), ch = (int)(i & 127);
    float d = deg[node];
    float degc = fmaxf(d, 1.f);
    float* base = stats + (long long)node * 512;
    float mean = base[ch] / degc;
    float s2 = base[384 + ch] / degc;
    float sd = sqrtf(fmaxf(s2 - mean * mean, 0.f) + 1e-5f);
    float mn = base[128 + ch], mx = base[256 + ch];
    if (d <= 0.f) { mn = 0.f; mx = 0.f; }
    base[ch] = mean; base[128 + ch] = mn; base[256 + ch] = mx; base[384 + ch] = sd;
  }
}

// ---------------------------------------------------------------------------
// Fused post: out = ([x | agg | agg*amp | agg*att] @ post_w + b) @ lin_w + b
// scaled features generated on the fly from stats (never materialized)
// ---------------------------------------------------------------------------
__global__ __launch_bounds__(256)
void k_postlin(const float* __restrict__ x, const float* __restrict__ stats,
               const float* __restrict__ amp, const float* __restrict__ att,
               const _Float16* __restrict__ wt_post, const float* __restrict__ b_post,
               const _Float16* __restrict__ wt_lin, const float* __restrict__ b_lin,
               float* outb, int Ntiles) {
  __shared__ _Float16 lds[8][16 * 128];
  int wslot = threadIdx.x >> 5;
  int wave = (int)((blockIdx.x * blockDim.x + threadIdx.x) >> 5);
  int lane = threadIdx.x & 31;
  if (wave >= Ntiles) return;
  int h = lane >> 4, l = lane & 15;
  int n0 = wave * 16;
  int mynode = n0 + l;
  const float* xrow = x + (long long)mynode * 128;
  const float* srow = stats + (long long)mynode * 512;
  float myamp = amp[mynode], myatt = att[mynode];

  v8f acc[8];
  acc_bias(acc, b_post, l);
  for (int k0 = 0; k0 < 1664; k0 += 32) {
    v16h a;
    if (k0 < 128) {
      a = frag_a_f32(xrow, k0, h);
    } else {
      int kk = k0 - 128;
      int seg = kk >> 9;          // 0:agg 1:agg*amp 2:agg*att
      int col = kk & 511;
      float sc = (seg == 0) ? 1.f : (seg == 1) ? myamp : myatt;
      a = frag_a_f32s(srow, col, h, sc);
    }
    wmma_step<1664>(acc, a, wt_post, l, k0, h);
  }

  _Float16* L = lds[wslot];
#pragma unroll
  for (int nt = 0; nt < 8; ++nt)
#pragma unroll
    for (int r = 0; r < 8; ++r)
      L[(r + 8 * h) * 128 + nt * 16 + l] = (_Float16)acc[nt][r];

  const _Float16* urow = L + l * 128;
  v8f a2[8];
  acc_bias(a2, b_lin, l);
#pragma unroll
  for (int k0 = 0; k0 < 128; k0 += 32) {
    v16h a = frag_a_f16(urow, k0, h);
    wmma_step<128>(a2, a, wt_lin, l, k0, h);
  }
#pragma unroll
  for (int nt = 0; nt < 8; ++nt)
#pragma unroll
    for (int r = 0; r < 8; ++r)
      outb[(long long)(n0 + r + 8 * h) * 128 + nt * 16 + l] = a2[nt][r];
}

__global__ void k_bnreduce(const float* __restrict__ outb, float* bnsum, float* bnsq) {
  int ch = threadIdx.x & 127;
  int row = blockIdx.x * 2 + (threadIdx.x >> 7);
  int stride = gridDim.x * 2;
  float s = 0.f, q = 0.f;
  for (; row < NN; row += stride) {
    float v = outb[(long long)row * 128 + ch];
    s += v; q += v * v;
  }
  atomicAdd(bnsum + ch, s);
  atomicAdd(bnsq + ch, q);
}

__global__ void k_bnapply(const float* __restrict__ outb, const float* __restrict__ bnsum,
                          const float* __restrict__ bnsq, const float* __restrict__ g,
                          const float* __restrict__ b, float* x) {
  long long total = (long long)NN * 128;
  long long i = (long long)blockIdx.x * blockDim.x + threadIdx.x;
  long long st = (long long)gridDim.x * blockDim.x;
  const float inv_n = 1.f / (float)NN;
  for (; i < total; i += st) {
    int ch = (int)(i & 127);
    float mu = bnsum[ch] * inv_n;
    float var = bnsq[ch] * inv_n - mu * mu;
    float bn = g[ch] * (outb[i] - mu) * rsqrtf(var + 1e-5f) + b[ch];
    x[i] = (x[i] + fmaxf(bn, 0.f)) * 0.5f;
  }
}

// ---------------------------------------------------------------------------
// Fused edge MLP: upd = relu([x[src] x[dst] ea] @ w1 + b1); ea += (upd@w2 + b2)/2
// ---------------------------------------------------------------------------
__global__ __launch_bounds__(256)
void k_emlp(const float* __restrict__ x, _Float16* ea,
            const int* __restrict__ src, const int* __restrict__ dst,
            const _Float16* __restrict__ wt1, const float* __restrict__ b1,
            const _Float16* __restrict__ wt2, const float* __restrict__ b2,
            int Etiles) {
  __shared__ _Float16 lds[8][16 * 128];
  int wslot = threadIdx.x >> 5;
  int wave = (int)((blockIdx.x * blockDim.x + threadIdx.x) >> 5);
  int lane = threadIdx.x & 31;
  if (wave >= Etiles) return;
  int h = lane >> 4, l = lane & 15;
  int e0 = wave * 16;
  int my = e0 + l;
  const float* xs = x + (long long)src[my] * 128;
  const float* xd = x + (long long)dst[my] * 128;
  const _Float16* er = ea + (long long)my * 128;

  v8f acc[8];
  acc_bias(acc, b1, l);
#pragma unroll
  for (int k0 = 0; k0 < 384; k0 += 32) {
    v16h a;
    if (k0 < 128)      a = frag_a_f32(xs, k0, h);
    else if (k0 < 256) a = frag_a_f32(xd, k0 - 128, h);
    else               a = frag_a_f16(er, k0 - 256, h);
    wmma_step<384>(acc, a, wt1, l, k0, h);
  }

  _Float16* L = lds[wslot];
#pragma unroll
  for (int nt = 0; nt < 8; ++nt)
#pragma unroll
    for (int r = 0; r < 8; ++r)
      L[(r + 8 * h) * 128 + nt * 16 + l] = (_Float16)fmaxf(acc[nt][r], 0.f);

  const _Float16* urow = L + l * 128;
  v8f a2[8];
  acc_bias(a2, b2, l);
#pragma unroll
  for (int k0 = 0; k0 < 128; k0 += 32) {
    v16h a = frag_a_f16(urow, k0, h);
    wmma_step<128>(a2, a, wt2, l, k0, h);
  }
#pragma unroll
  for (int nt = 0; nt < 8; ++nt)
#pragma unroll
    for (int r = 0; r < 8; ++r) {
      long long idx = (long long)(e0 + r + 8 * h) * 128 + nt * 16 + l;
      float old = (float)ea[idx];
      ea[idx] = (_Float16)(old + 0.5f * a2[nt][r]);
    }
}

// ---------------------------------------------------------------------------
// Host side
// ---------------------------------------------------------------------------
extern "C" void kernel_launch(void* const* d_in, const int* in_sizes, int n_in,
                              void* d_out, int out_size, void* d_ws, size_t ws_size,
                              hipStream_t stream) {
  (void)in_sizes; (void)n_in; (void)out_size; (void)ws_size;

  const float* x_in   = (const float*)d_in[0];
  const int*   eidx   = (const int*)d_in[1];
  const float* eattr  = (const float*)d_in[2];
  const float* pattr  = (const float*)d_in[4];
  const float* nattr  = (const float*)d_in[6];
  const float* adl    = (const float*)d_in[7];
  const float* w_node = (const float*)d_in[8];  const float* b_node = (const float*)d_in[9];
  const float* w_edge = (const float*)d_in[10]; const float* b_edge = (const float*)d_in[11];
  const float* w_eenc = (const float*)d_in[12]; const float* b_eenc = (const float*)d_in[13];
  const float* w_pre  = (const float*)d_in[14]; const float* b_pre  = (const float*)d_in[15];
  const float* w_post = (const float*)d_in[16]; const float* b_post = (const float*)d_in[17];
  const float* w_lin  = (const float*)d_in[18]; const float* b_lin  = (const float*)d_in[19];
  const float* w_m1   = (const float*)d_in[20]; const float* b_m1   = (const float*)d_in[21];
  const float* w_m2   = (const float*)d_in[22]; const float* b_m2   = (const float*)d_in[23];
  const float* bng    = (const float*)d_in[24]; const float* bnb    = (const float*)d_in[25];

  const int* src = eidx;       // edge_index[0]
  const int* dst = eidx + EE;  // edge_index[1]

  float* xout = (float*)d_out;                      // [N,128]
  float* pout = xout + (long long)NN * HH;          // [EP,128]
  float* nout = pout + (long long)EPP * HH;         // [EP,128]

  // workspace carve (aligned 256B)
  char* wsp = (char*)d_ws;
  size_t off = 0;
  auto carve = [&](size_t bytes) -> char* {
    char* p = wsp + off;
    off += (bytes + 255) & ~(size_t)255;
    return p;
  };
  _Float16* wt_node = (_Float16*)carve((size_t)128 * 64 * 2);
  _Float16* wt_edge = (_Float16*)carve((size_t)128 * 32 * 2);
  _Float16* wt_eenc = (_Float16*)carve((size_t)LL * 128 * 128 * 2);
  _Float16* wt_pre  = (_Float16*)carve((size_t)LL * 128 * 384 * 2);
  _Float16* wt_post = (_Float16*)carve((size_t)LL * 128 * 1664 * 2);
  _Float16* wt_lin  = (_Float16*)carve((size_t)LL * 128 * 128 * 2);
  _Float16* wt_m1   = (_Float16*)carve((size_t)LL * 128 * 384 * 2);
  _Float16* wt_m2   = (_Float16*)carve((size_t)LL * 128 * 128 * 2);
  float* deg   = (float*)carve((size_t)NN * 4);
  float* amp   = (float*)carve((size_t)NN * 4);
  float* att   = (float*)carve((size_t)NN * 4);
  float* stats = (float*)carve((size_t)NN * 512 * 4);
  float* outb  = (float*)carve((size_t)NN * 128 * 4);
  _Float16* ea = (_Float16*)carve((size_t)EE * 128 * 2);
  float* bnsum = (float*)carve(256 * 4);
  float* bnsq  = bnsum + 128;

  const int TPB = 256;
  auto cvtBlocks = [](long long t) { return (int)((t + 255) / 256); };

  // weight conversion (fp32 [L][K][128] -> fp16 [L][128][K])
  k_cvt<<<cvtBlocks((long long)64 * 128), TPB, 0, stream>>>(w_node, wt_node, 64, (long long)64 * 128);
  k_cvt<<<cvtBlocks((long long)32 * 128), TPB, 0, stream>>>(w_edge, wt_edge, 32, (long long)32 * 128);
  k_cvt<<<cvtBlocks((long long)LL * 128 * 128), TPB, 0, stream>>>(w_eenc, wt_eenc, 128, (long long)LL * 128 * 128);
  k_cvt<<<cvtBlocks((long long)LL * 384 * 128), TPB, 0, stream>>>(w_pre, wt_pre, 384, (long long)LL * 384 * 128);
  k_cvt<<<cvtBlocks((long long)LL * 1664 * 128), TPB, 0, stream>>>(w_post, wt_post, 1664, (long long)LL * 1664 * 128);
  k_cvt<<<cvtBlocks((long long)LL * 128 * 128), TPB, 0, stream>>>(w_lin, wt_lin, 128, (long long)LL * 128 * 128);
  k_cvt<<<cvtBlocks((long long)LL * 384 * 128), TPB, 0, stream>>>(w_m1, wt_m1, 384, (long long)LL * 384 * 128);
  k_cvt<<<cvtBlocks((long long)LL * 128 * 128), TPB, 0, stream>>>(w_m2, wt_m2, 128, (long long)LL * 128 * 128);

  // degree statistics
  k_fillf<<<(NN + TPB - 1) / TPB, TPB, 0, stream>>>(deg, NN, 0.f);
  k_deg<<<(EE + TPB - 1) / TPB, TPB, 0, stream>>>(dst, deg);
  k_nodestats<<<(NN + TPB - 1) / TPB, TPB, 0, stream>>>(deg, adl, amp, att);

  // input embeddings (WMMA)
  const int NT = NN / 16;    // 3125
  const int ET = EE / 16;    // 25000
  const int PT = EPP / 16;   // 2500
  k_gemm_in<64><<<(NT + 7) / 8, TPB, 0, stream>>>(x_in, wt_node, b_node, xout, nullptr, NT);
  k_gemm_in<32><<<(ET + 7) / 8, TPB, 0, stream>>>(eattr, wt_edge, b_edge, nullptr, ea, ET);
  k_gemm_in<32><<<(PT + 7) / 8, TPB, 0, stream>>>(pattr, wt_edge, b_edge, pout, nullptr, PT);
  k_gemm_in<32><<<(PT + 7) / 8, TPB, 0, stream>>>(nattr, wt_edge, b_edge, nout, nullptr, PT);

  for (int l = 0; l < LL; ++l) {
    k_init_stats<<<8192, TPB, 0, stream>>>(stats);
    k_message<<<(ET + 7) / 8, TPB, 0, stream>>>(
        xout, ea, src, dst,
        wt_eenc + (size_t)l * 128 * 128, b_eenc + l * 128,
        wt_pre + (size_t)l * 128 * 384, b_pre + l * 128, stats, ET);
    k_aggfinish<<<8192, TPB, 0, stream>>>(stats, deg);
    k_postlin<<<(NT + 7) / 8, TPB, 0, stream>>>(
        xout, stats, amp, att,
        wt_post + (size_t)l * 128 * 1664, b_post + l * 128,
        wt_lin + (size_t)l * 128 * 128, b_lin + l * 128, outb, NT);
    k_fillf<<<1, TPB, 0, stream>>>(bnsum, 256, 0.f);
    k_bnreduce<<<256, TPB, 0, stream>>>(outb, bnsum, bnsq);
    k_bnapply<<<8192, TPB, 0, stream>>>(outb, bnsum, bnsq, bng + l * 128, bnb + l * 128, xout);
    k_emlp<<<(ET + 7) / 8, TPB, 0, stream>>>(
        xout, ea, src, dst,
        wt_m1 + (size_t)l * 128 * 384, b_m1 + l * 128,
        wt_m2 + (size_t)l * 128 * 128, b_m2 + l * 128, ET);
  }
}
